// SmallLSTMNet_22162031247994
// MI455X (gfx1250) — compile-verified, matrix operand order checked by reading, not run on hardware
//
#include <hip/hip_runtime.h>

// Problem constants (match reference)
#define BB 4096
#define TT 512
#define II 5
#define HH 10
#define GG 40   // 4*H

// LDS strides (words), chosen for conflict-free bank access
#define ROWG 52   // gate buffer row stride: 52*b mod 64 distinct over b=0..15, halves disjoint mod 4
#define HSTR 12   // h buffer row stride

typedef __bf16 bf16_t;
typedef bf16_t v16bf __attribute__((ext_vector_type(16)));
typedef float  v8f   __attribute__((ext_vector_type(8)));

union BfTile { unsigned u[8]; v16bf v; };

// pack two f32 into one dword of bf16 (truncating): low16 = bf16(lo), high16 = bf16(hi)
__device__ __forceinline__ unsigned pk(float lo, float hi) {
  return __builtin_amdgcn_perm(__float_as_uint(hi), __float_as_uint(lo), 0x07060302u);
}
__device__ __forceinline__ float sigm(float x) {
  return __builtin_amdgcn_rcpf(1.0f + __expf(-x));          // v_exp_f32 + v_rcp_f32
}
__device__ __forceinline__ float tanh_f(float x) {
  return 1.0f - 2.0f * __builtin_amdgcn_rcpf(1.0f + __expf(2.0f * x));
}

// Single-wave workgroup: LDS ordering only needs a DS-counter wait (DS ops are
// in-order within a wave); crucially this does NOT wait on LOADcnt, so the
// double-buffered global x loads stay off the critical path (unlike
// __syncthreads(), which lowers to s_wait_loadcnt_dscnt 0x0).
__device__ __forceinline__ void lds_fence() {
  asm volatile("s_wait_dscnt 0x0" ::: "memory");
}

__global__ __launch_bounds__(32)
void lstm_wmma_kernel(const float* __restrict__ x,
                      const float* __restrict__ W_ih,
                      const float* __restrict__ W_hh,
                      const float* __restrict__ b_ih,
                      const float* __restrict__ b_hh,
                      const float* __restrict__ W_fc,
                      const float* __restrict__ b_fc,
                      float* __restrict__ out) {
  __shared__ float gbuf[16 * ROWG];   // gates, f32, row = batch-in-tile, col = gate 0..47
  __shared__ float hbuf[16 * HSTR];   // hidden state, f32, row = batch-in-tile

  const int lane  = threadIdx.x;
  const int laneN = lane & 15;        // N column / batch-in-tile for elementwise
  const int half  = lane >> 4;        // lane half: A/B K-layout + elementwise j-half
  const int bbase = blockIdx.x * 16;  // 16 batch rows per wave -> 256 waves

  // zero hidden-state buffer (h0 = 0)
  #pragma unroll
  for (int k = 0; k < 6; ++k) hbuf[lane + 32 * k] = 0.0f;

  // ---- Build loop-invariant B operand tiles: K = [W_hh rows 0..9 | W_ih rows 0..4 | 0],
  //      N = gate. bf16 16x32 B layout: lanes 0-15 hold K=0..7 (pairs in V0..V3),
  //      lanes 16-31 hold K=8..15 in V0..V3; K=16..31 (V4..V7) = 0.
  BfTile Bt[3];
  float  biasv[3];
  #pragma unroll
  for (int tle = 0; tle < 3; ++tle) {
    int  g     = tle * 16 + laneN;
    bool valid = (g < GG);
    int  gg    = valid ? g : (GG - 1);
    const float* wh = W_hh + gg * HH;  // W_hh[g][k], k=0..9
    const float* wi = W_ih + gg * II;  // W_ih[g][k], k=0..4
    unsigned u0, u1, u2, u3;
    if (half == 0) {
      u0 = pk(wh[0], wh[1]); u1 = pk(wh[2], wh[3]);
      u2 = pk(wh[4], wh[5]); u3 = pk(wh[6], wh[7]);
    } else {
      u0 = pk(wh[8], wh[9]); u1 = pk(wi[0], wi[1]);
      u2 = pk(wi[2], wi[3]); u3 = pk(wi[4], 0.0f);
    }
    Bt[tle].u[0] = valid ? u0 : 0u;
    Bt[tle].u[1] = valid ? u1 : 0u;
    Bt[tle].u[2] = valid ? u2 : 0u;
    Bt[tle].u[3] = valid ? u3 : 0u;
    Bt[tle].u[4] = 0u; Bt[tle].u[5] = 0u; Bt[tle].u[6] = 0u; Bt[tle].u[7] = 0u;
    biasv[tle]   = valid ? (b_ih[g] + b_hh[g]) : 0.0f;  // bias rides in C accumulator
  }

  // per-lane recurrent state: (batch = laneN, hidden j = half*5 + jj)
  float c0 = 0, c1 = 0, c2 = 0, c3 = 0, c4 = 0;
  float h0 = 0, h1 = 0, h2 = 0, h3 = 0, h4 = 0;

  // x row for this lane's batch; prime the register double-buffer with t=0
  const float* xrow = x + (size_t)(bbase + laneN) * TT * II;
  float xc0 = xrow[0], xc1 = xrow[1], xc2 = xrow[2], xc3 = xrow[3], xc4 = xrow[4];

  lds_fence();   // h-buffer zeros visible before first A-build

  const int hoff  = laneN * HSTR + (half ? 8 : 0);
  const int hoff2 = laneN * HSTR + (half ? 8 : 4);

  #pragma unroll 1
  for (int t = 0; t < TT; ++t) {
    // ---- overlap: kick off next-step x loads + prefetch ahead (global_prefetch_b8)
    int tn = (t + 1 < TT) ? (t + 1) : t;
    const float* xr = xrow + tn * II;
    float xn0 = xr[0], xn1 = xr[1], xn2 = xr[2], xn3 = xr[3], xn4 = xr[4];
    int tp = (t + 16 < TT) ? (t + 16) : t;
    __builtin_prefetch(xrow + tp * II, 0, 1);

    // ---- Build A operand (bf16 16x32): M=batch, K=[h(10)|x(5)|0-pad]
    //   lanes 0-15:  V0..V3 = (h0,h1)(h2,h3)(h4,h5)(h6,h7)
    //   lanes 16-31: V0 = (h8,h9), V1..V3 = (x0,x1)(x2,x3)(x4,0)
    float4 ha = *(const float4*)(&hbuf[hoff]);
    float4 hb = *(const float4*)(&hbuf[hoff2]);
    BfTile A;
    A.u[0] = pk(ha.x, ha.y);                               // (h0,h1) / (h8,h9)
    A.u[1] = half ? pk(xc0, xc1) : pk(ha.z, ha.w);
    A.u[2] = half ? pk(xc2, xc3) : pk(hb.x, hb.y);
    A.u[3] = half ? pk(xc4, 0.0f) : pk(hb.z, hb.w);
    A.u[4] = 0u; A.u[5] = 0u; A.u[6] = 0u; A.u[7] = 0u;

    // ---- 3 WMMAs (one per 16-gate tile), C pre-loaded with bias; scatter D -> gbuf
    #pragma unroll
    for (int tle = 0; tle < 3; ++tle) {
      float bv = biasv[tle];
      v8f cc;
      cc[0] = bv; cc[1] = bv; cc[2] = bv; cc[3] = bv;
      cc[4] = bv; cc[5] = bv; cc[6] = bv; cc[7] = bv;
      cc = __builtin_amdgcn_wmma_f32_16x16x32_bf16(
               false, A.v, false, Bt[tle].v, (short)0, cc, false, false);
      int colbase = tle * 16 + laneN;
      #pragma unroll
      for (int vv = 0; vv < 8; ++vv)   // D layout: lane=N, VGPR=M (+8 for upper half)
        gbuf[(half * 8 + vv) * ROWG + colbase] = cc[vv];
    }
    lds_fence();   // gates visible to elementwise gathers (DS-only wait)

    // ---- elementwise LSTM cell: lane handles (batch laneN, j = half*5 + jj)
    const float* grow = &gbuf[laneN * ROWG + half * 5];
    float* hrow = &hbuf[laneN * HSTR + half * 5];
    #define CELL(jj, cR, hR)                                                    \
      { float ig = grow[jj];      float fg = grow[10 + jj];                     \
        float g2 = grow[20 + jj]; float og = grow[30 + jj];                     \
        float iv = sigm(ig), fv = sigm(fg), gv = tanh_f(g2), ov = sigm(og);     \
        cR = fv * cR + iv * gv;                                                 \
        hR = ov * tanh_f(cR);                                                   \
        hrow[jj] = hR; }
    CELL(0, c0, h0) CELL(1, c1, h1) CELL(2, c2, h2) CELL(3, c3, h3) CELL(4, c4, h4)
    #undef CELL
    lds_fence();   // h visible to next step's A-build (DS-only wait)

    xc0 = xn0; xc1 = xn1; xc2 = xn2; xc3 = xn3; xc4 = xn4;
  }

  // ---- final fc + tanh: dot(h, W_fc) split across lane halves, fold with shfl_xor
  const float* wf = W_fc + half * 5;
  float partial = h0 * wf[0] + h1 * wf[1] + h2 * wf[2] + h3 * wf[3] + h4 * wf[4];
  partial += __shfl_xor(partial, 16);
  if (half == 0) out[bbase + laneN] = tanh_f(partial + b_fc[0]);
}

extern "C" void kernel_launch(void* const* d_in, const int* in_sizes, int n_in,
                              void* d_out, int out_size, void* d_ws, size_t ws_size,
                              hipStream_t stream) {
  const float* x   = (const float*)d_in[0];
  const float* Wih = (const float*)d_in[1];
  const float* Whh = (const float*)d_in[2];
  const float* bih = (const float*)d_in[3];
  const float* bhh = (const float*)d_in[4];
  const float* Wfc = (const float*)d_in[5];
  const float* bfc = (const float*)d_in[6];
  float* out = (float*)d_out;
  (void)in_sizes; (void)n_in; (void)out_size; (void)d_ws; (void)ws_size;

  dim3 grid(BB / 16), block(32);   // 256 single-wave workgroups, 16 batch rows each
  hipLaunchKernelGGL(lstm_wmma_kernel, grid, block, 0, stream,
                     x, Wih, Whh, bih, bhh, Wfc, bfc, out);
}
